// WordFilterSelfAttention_61280593379536
// MI455X (gfx1250) — compile-verified, matrix operand order checked by reading, not run on hardware
//
#include <hip/hip_runtime.h>
#include <math.h>

// ---------------------------------------------------------------------------
// WordFilterSelfAttention for MI455X (gfx1250, wave32, WMMA bf16 path)
//
// scores = W2^T . tanh(word_out @ W1 + b1) + b2   (h never materialized)
// Dominant GEMM: [25600 x 768] @ [768 x 768] = 30.2 GFLOP in bf16 WMMA.
// A tile (64 rows) staged once in LDS; W1 pre-packed to bf16 fragments in L2.
// ---------------------------------------------------------------------------

typedef __attribute__((ext_vector_type(16))) __bf16 v16bf;
typedef __attribute__((ext_vector_type(8)))  __bf16 v8bf;
typedef __attribute__((ext_vector_type(8)))  float  v8f;

#define BB 8
#define SS 64
#define TT 50
#define DD 768
#define KCH (DD/32)      // 24 K-chunks of 32
#define NCH (DD/16)      // 48 N-chunks of 16
#define MTOT (BB*SS*TT)  // 25600
#define NEGV (-1.0e9f)

// ---------------------------------------------------------------------------
// Kernel 1: convert W1 (f32 row-major [k][n]) into bf16 WMMA B-fragment order.
// Layout: [nChunk 0..47][kChunk 0..23][lane 0..31][16 bf16]  (32 B per lane)
//   lane = (n%16) + 16*((k%16) >= 8), pos = (k%8) + 8*((k%32)/16)
// ---------------------------------------------------------------------------
__global__ void prep_w1_kernel(const float* __restrict__ W1,
                               __bf16* __restrict__ wfrag) {
  int t = blockIdx.x * blockDim.x + threadIdx.x;   // 48*24*32*2 = 73728 items of 8
  if (t >= NCH * KCH * 32 * 2) return;
  int hi   = t & 1;
  int lane = (t >> 1) & 31;
  int kc   = (t >> 6) % KCH;
  int nc   = t / (64 * KCH);
  int nl = lane & 15, lh = lane >> 4;
  int n = nc * 16 + nl;
  __bf16* dst = wfrag + (((size_t)(nc * KCH + kc) * 32 + lane) * 16 + hi * 8);
#pragma unroll
  for (int p = 0; p < 8; ++p) {
    int kk = hi * 16 + lh * 8 + p;
    int k = kc * 32 + kk;
    dst[p] = (__bf16)W1[(size_t)k * DD + n];
  }
}

// ---------------------------------------------------------------------------
// Kernel 2: fused GEMM + tanh + W2-reduction -> raw scores[25600].
// 256 threads (8 waves), 64 rows per workgroup, 400 workgroups.
// Uniform 6-iteration N loop with a trailing __syncthreads(): the barrier is
// an LDS fence, so the compiler cannot hoist the (j-invariant) A-fragment
// ds loads out of the loop -> no VGPR blow-up, no scratch spills.
// ---------------------------------------------------------------------------
__global__ __launch_bounds__(256) void gemm_scores_kernel(
    const float* __restrict__ wordOut, const __bf16* __restrict__ wfrag,
    const float* __restrict__ b1, const float* __restrict__ W2,
    float* __restrict__ rawScores) {
  __shared__ __bf16 Ash[KCH * 4 * 32 * 16];  // 98304 B, fragment-ordered A tile
  __shared__ float scoreBuf[64];

  const int tid = threadIdx.x;
  const int row0 = blockIdx.x * 64;

  // ---- stage A (f32 -> bf16) in WMMA fragment order; 8 consecutive k per
  //      item land contiguously -> single 16 B ds_store ----
  for (int it = tid; it < 64 * 96; it += 256) {
    int m = it / 96, g = it % 96;                 // g = k/8
    const float* src = wordOut + (size_t)(row0 + m) * DD + g * 8;
    float4 f0 = *(const float4*)src;
    float4 f1 = *(const float4*)(src + 4);
    int kc = g >> 2, q = g & 3;
    int lane = (m & 15) + ((q & 1) << 4);
    int mg = m >> 4;
    int pos = (q >> 1) << 3;
    v8bf v = {(__bf16)f0.x, (__bf16)f0.y, (__bf16)f0.z, (__bf16)f0.w,
              (__bf16)f1.x, (__bf16)f1.y, (__bf16)f1.z, (__bf16)f1.w};
    *(v8bf*)(Ash + (((kc * 4 + mg) * 32 + lane) * 16 + pos)) = v;
  }
  if (tid < 64) scoreBuf[tid] = 0.f;
  __syncthreads();

  const int wave = tid >> 5;
  const int lane = tid & 31;
  float sacc[4][8];
#pragma unroll
  for (int g = 0; g < 4; ++g)
#pragma unroll
    for (int r = 0; r < 8; ++r) sacc[g][r] = 0.f;

#pragma unroll 1
  for (int j = 0; j < NCH / 8; ++j) {   // uniform trip count for ALL waves
    const int nc = j * 8 + wave;
    const int col = nc * 16 + (lane & 15);
    const float b1v = b1[col];
    const float w2v = W2[col];
    v8f acc0 = {}, acc1 = {}, acc2 = {}, acc3 = {};
    const __bf16* bptr = wfrag + ((size_t)nc * KCH) * 512 + lane * 16;
    const __bf16* aptr = Ash + lane * 16;
#pragma unroll 1
    for (int kc = 0; kc < KCH; ++kc) {
      v16bf bf = *(const v16bf*)bptr;
      v16bf a0 = *(const v16bf*)(aptr);
      v16bf a1 = *(const v16bf*)(aptr + 512);
      v16bf a2 = *(const v16bf*)(aptr + 1024);
      v16bf a3 = *(const v16bf*)(aptr + 1536);
      bptr += 512;
      aptr += 2048;
      acc0 = __builtin_amdgcn_wmma_f32_16x16x32_bf16(false, a0, false, bf, (short)0, acc0, false, false);
      acc1 = __builtin_amdgcn_wmma_f32_16x16x32_bf16(false, a1, false, bf, (short)0, acc1, false, false);
      acc2 = __builtin_amdgcn_wmma_f32_16x16x32_bf16(false, a2, false, bf, (short)0, acc2, false, false);
      acc3 = __builtin_amdgcn_wmma_f32_16x16x32_bf16(false, a3, false, bf, (short)0, acc3, false, false);
    }
    // fused epilogue: h = tanh(acc + b1[n]); score += h * W2[n]
#pragma unroll
    for (int r = 0; r < 8; ++r) {
      sacc[0][r] += tanhf(acc0[r] + b1v) * w2v;
      sacc[1][r] += tanhf(acc1[r] + b1v) * w2v;
      sacc[2][r] += tanhf(acc2[r] + b1v) * w2v;
      sacc[3][r] += tanhf(acc3[r] + b1v) * w2v;
    }
    __syncthreads();   // LDS fence: blocks LICM hoist of A loads across j
  }

  // reduce across the 16 lanes holding one row's columns, then across waves
#pragma unroll
  for (int g = 0; g < 4; ++g) {
#pragma unroll
    for (int r = 0; r < 8; ++r) {
      float v = sacc[g][r];
      v += __shfl_xor(v, 1, 32);
      v += __shfl_xor(v, 2, 32);
      v += __shfl_xor(v, 4, 32);
      v += __shfl_xor(v, 8, 32);
      if ((lane & 15) == 0)   // lane 0 -> rows g*16+r ; lane 16 -> rows g*16+8+r
        atomicAdd(&scoreBuf[g * 16 + ((lane >> 4) << 3) + r], v);
    }
  }
  __syncthreads();
  if (tid < 64) rawScores[row0 + tid] = scoreBuf[tid];
}

// ---------------------------------------------------------------------------
// Kernel 3: per (b,s): mask (+b2), softmax over T=50, stable top-5, keep mask,
// topk indices (as float), and gather of the 5x768 top word vectors.
// ---------------------------------------------------------------------------
__global__ __launch_bounds__(256) void finish_kernel(
    const float* __restrict__ wordOut, const long long* __restrict__ x,
    const float* __restrict__ b2, const float* __restrict__ rawScores,
    float* __restrict__ out) {
  const int bs = blockIdx.x;     // 0..511
  const int tid = threadIdx.x;
  __shared__ float sc[TT];
  __shared__ float ev[TT];
  __shared__ float keepf[TT];
  __shared__ int topki[5];
  __shared__ float red[2];

  float* scoresOut = out + (size_t)BB * SS * TT * DD;
  float* keepOut = scoresOut + BB * SS * TT;
  float* attnOut = keepOut + BB * SS * TT;
  float* topkOut = attnOut + BB * SS * TT;
  float* tvecOut = topkOut + BB * SS * 5;

  bool pad = false;
  if (tid < TT) {
    pad = (x[(size_t)bs * TT + tid] == 0ll);
    float s = rawScores[bs * TT + tid] + b2[0];
    s = pad ? NEGV : s;
    sc[tid] = s;
    keepf[tid] = 0.f;
    scoresOut[bs * TT + tid] = s;
  }
  __syncthreads();
  if (tid == 0) {
    float mx = sc[0];
    for (int t = 1; t < TT; ++t) mx = fmaxf(mx, sc[t]);
    red[0] = mx;
  }
  __syncthreads();
  if (tid < TT) ev[tid] = expf(sc[tid] - red[0]);
  __syncthreads();
  if (tid == 0) {
    float s = 0.f;
    for (int t = 0; t < TT; ++t) s += ev[t];
    red[1] = s;
    // stable top-5: strict '>' so ties keep the lowest index (matches lax.top_k)
    for (int i = 0; i < 5; ++i) {
      float best = -INFINITY;
      int bi = 0;
      for (int t = 0; t < TT; ++t)
        if (keepf[t] == 0.f && sc[t] > best) { best = sc[t]; bi = t; }
      keepf[bi] = 1.f;
      topki[i] = bi;
    }
  }
  __syncthreads();
  if (tid < TT) {
    attnOut[bs * TT + tid] = ev[tid] / red[1];
    keepOut[bs * TT + tid] = pad ? 0.f : keepf[tid];
  }
  if (tid < 5) topkOut[bs * 5 + tid] = (float)topki[tid];
  if (tid < DD / 4) {   // 192 lanes copy float4s; 5 rows of 768 floats
#pragma unroll
    for (int i = 0; i < 5; ++i) {
      const float4* src = (const float4*)(wordOut + ((size_t)bs * TT + topki[i]) * DD);
      float4* dst = (float4*)(tvecOut + ((size_t)bs * 5 + i) * DD);
      dst[tid] = src[tid];
    }
  }
}

// ---------------------------------------------------------------------------
extern "C" void kernel_launch(void* const* d_in, const int* in_sizes, int n_in,
                              void* d_out, int out_size, void* d_ws, size_t ws_size,
                              hipStream_t stream) {
  const float* wordOut = (const float*)d_in[0];
  const long long* x   = (const long long*)d_in[1];
  const float* W1      = (const float*)d_in[2];
  const float* b1      = (const float*)d_in[3];
  const float* W2      = (const float*)d_in[4];
  const float* b2      = (const float*)d_in[5];
  float* out = (float*)d_out;

  // workspace: [bf16 W1 fragments 1.18 MB][raw scores 100 KB]
  __bf16* wfrag = (__bf16*)d_ws;
  float* rawScores = (float*)((char*)d_ws + (size_t)DD * DD * sizeof(__bf16));

  // filtered_word_out == word_out exactly (km in {0,1}) -> plain D2D copy
  hipMemcpyAsync(d_out, d_in[0], (size_t)BB * SS * TT * DD * sizeof(float),
                 hipMemcpyDeviceToDevice, stream);

  int prepItems = NCH * KCH * 32 * 2;  // 73728
  prep_w1_kernel<<<(prepItems + 255) / 256, 256, 0, stream>>>(W1, wfrag);
  gemm_scores_kernel<<<MTOT / 64, 256, 0, stream>>>(wordOut, wfrag, b1, W2, rawScores);
  finish_kernel<<<BB * SS, 256, 0, stream>>>(wordOut, x, b2, rawScores, out);
}
